// CRD_61100204753128
// MI455X (gfx1250) — compile-verified, matrix operand order checked by reading, not run on hardware
//
#include <hip/hip_runtime.h>
#include <hip/hip_bf16.h>

#define DFEAT 128
#define TILE_ROWS 16
#define ROWTILES 4                       // 4 x 16 = 64 rows per block

typedef __attribute__((ext_vector_type(16))) _Float16 v16h;
typedef __attribute__((ext_vector_type(8)))  _Float16 v8h;
typedef __attribute__((ext_vector_type(8)))  float    v8f;
typedef __attribute__((ext_vector_type(4)))  int      v4i;

// ---------------- degree / norm ----------------

__global__ void crd_deg_init(float* deg, int n) {
    int i = blockIdx.x * blockDim.x + threadIdx.x;
    if (i < n) deg[i] = 1.0f;                 // self-loop contributes 1
}

__global__ void crd_deg_edges(const int* __restrict__ dst, float* deg, int e) {
    int i = blockIdx.x * blockDim.x + threadIdx.x;
    if (i < e) atomicAdd(&deg[dst[i]], 1.0f);
}

__global__ void crd_dinv(float* deg, int n) {
    int i = blockIdx.x * blockDim.x + threadIdx.x;
    if (i < n) deg[i] = rsqrtf(deg[i]);       // deg >= 1 always (self-loop)
}

// ---------------- one-shot: W (f32, row-major [k][n]) -> Wt (f16, [n][k]) ----------------

__global__ void crd_prep_w(const float* __restrict__ w, _Float16* __restrict__ wT) {
    int i = blockIdx.x * blockDim.x + threadIdx.x;
    if (i < DFEAT * DFEAT) {
        int ncol = i >> 7, k = i & 127;
        wT[i] = (_Float16)w[k * DFEAT + ncol];
    }
}

// ---------------- GEMM: h = x @ W via WMMA f16->f32 ----------------
// Block = 256 threads = 8 waves, handles 64 rows x 128 cols.
// Wave w owns column tile w and reuses its B fragments across 4 row tiles.

__global__ void crd_gemm(const float* __restrict__ x,
                         const _Float16* __restrict__ wT,   // [n][k] f16, pre-transposed
                         float* __restrict__ h, int n) {
    __shared__ _Float16 wTs[DFEAT * DFEAT];                 // 32 KB
    __shared__ _Float16 xs[ROWTILES * TILE_ROWS * DFEAT];   // 64x128 f16 = 16 KB

    const int t = threadIdx.x;
    const int row0 = blockIdx.x * (ROWTILES * TILE_ROWS);

    // ---- Wt (32KB f16) -> LDS, via gfx1250 async copy when available ----
#if __has_builtin(__builtin_amdgcn_global_load_async_to_lds_b128)
    {
        const char* gsrc = (const char*)wT + t * 16;
        char*       ldst = (char*)wTs + t * 16;
        #pragma unroll
        for (int i = 0; i < 8; ++i) {           // 8 x 16B per thread = 32KB total
            __builtin_amdgcn_global_load_async_to_lds_b128(
                (__attribute__((address_space(1))) v4i*)(gsrc + i * 4096),
                (__attribute__((address_space(3))) v4i*)(ldst + i * 4096),
                0, 0);
        }
#if __has_builtin(__builtin_amdgcn_s_wait_asynccnt)
        __builtin_amdgcn_s_wait_asynccnt(0);
#else
        asm volatile("s_wait_asynccnt 0" ::: "memory");
#endif
    }
#else
    for (int i = t * 8; i < DFEAT * DFEAT; i += 256 * 8)
        *(v8h*)&wTs[i] = *(const v8h*)&wT[i];
#endif

    // ---- x tile (up to 64 rows f32) -> f16 LDS ----
    const int elems = ROWTILES * TILE_ROWS * DFEAT;         // 8192
    int limit = (n - row0) * DFEAT;
    if (limit > elems) limit = elems;                       // limit % 4 == 0 (DFEAT mult of 4)
    for (int i = t * 4; i < elems; i += 256 * 4) {
        float4 v = make_float4(0.f, 0.f, 0.f, 0.f);
        if (i < limit) v = *(const float4*)&x[(size_t)row0 * DFEAT + i];
        _Float16* o = &xs[i];
        o[0] = (_Float16)v.x; o[1] = (_Float16)v.y;
        o[2] = (_Float16)v.z; o[3] = (_Float16)v.w;
    }
    __syncthreads();

    const int wave = t >> 5;          // 0..7: output column tile
    const int lane = t & 31;
    const int l16  = lane & 15;       // M (A/C) or N (B/C)
    const int g    = lane >> 4;       // K sub-range / M half selector
    const int colBase = wave * 16;

    // ---- B fragments, built once, reused for all 4 row tiles ----
    // elem i<8: K = kk*32 + 8g + i ; elem i>=8: K = kk*32 + 16 + 8g + (i-8)
    v16h bfrag[4];
    #pragma unroll
    for (int kk = 0; kk < 4; ++kk) {
        const int kA = kk * 32 + 8 * g;
        v8h blo = *(const v8h*)(&wTs[(colBase + l16) * DFEAT + kA]);
        v8h bhi = *(const v8h*)(&wTs[(colBase + l16) * DFEAT + kA + 16]);
        #pragma unroll
        for (int i = 0; i < 8; ++i) { bfrag[kk][i] = blo[i]; bfrag[kk][i + 8] = bhi[i]; }
    }

    // ---- 4 row tiles x 4 K-steps ----
    #pragma unroll
    for (int rt = 0; rt < ROWTILES; ++rt) {
        const int mrow = rt * TILE_ROWS;
        const int grow0 = row0 + mrow;
        if (grow0 >= n) break;

        v8f c = {};
        #pragma unroll
        for (int kk = 0; kk < 4; ++kk) {
            const int kA = kk * 32 + 8 * g;
            v8h alo = *(const v8h*)(&xs[(mrow + l16) * DFEAT + kA]);
            v8h ahi = *(const v8h*)(&xs[(mrow + l16) * DFEAT + kA + 16]);
            v16h a;
            #pragma unroll
            for (int i = 0; i < 8; ++i) { a[i] = alo[i]; a[i + 8] = ahi[i]; }
            c = __builtin_amdgcn_wmma_f32_16x16x32_f16(
                    false, a, false, bfrag[kk], (short)0, c, false, false);
        }

        // C/D layout: lane l16 = N, VGPR r -> M = r + 8*g
        if (grow0 + TILE_ROWS <= n) {           // full tile: straight-line stores
            #pragma unroll
            for (int r = 0; r < 8; ++r)
                h[(size_t)(grow0 + r + 8 * g) * DFEAT + colBase + l16] = c[r];
        } else {
            int rows_here = n - grow0;
            #pragma unroll
            for (int r = 0; r < 8; ++r) {
                int m = r + 8 * g;
                if (m < rows_here)
                    h[(size_t)(grow0 + m) * DFEAT + colBase + l16] = c[r];
            }
        }
    }
}

// ---------------- out init: self-loop term out[i] = h[i] * dinv[i]^2 ----------------

__global__ void crd_out_init(const float* __restrict__ h,
                             const float* __restrict__ dinv,
                             float* __restrict__ out, int n) {
    int idx = blockIdx.x * blockDim.x + threadIdx.x;
    if (idx < n * DFEAT) {
        int node = idx >> 7;
        float dv = dinv[node];
        out[idx] = h[idx] * dv * dv;
    }
}

// ---------------- edge aggregation: out[dst] += dinv[src]*dinv[dst] * h[src] ----------------
// One wave per edge; lane loads float4 of h (coalesced 512B/row) + 4 f32 atomics.
// h and out both fit in the 192MB L2, so gathers and atomics resolve in L2.

__global__ void crd_edge_agg(const int* __restrict__ src,
                             const int* __restrict__ dst,
                             const float* __restrict__ dinv,
                             const float* __restrict__ h,
                             float* out, int e) {
    int tid  = blockIdx.x * blockDim.x + threadIdx.x;
    int edge = tid >> 5;
    int lane = tid & 31;
    if (edge >= e) return;
    int s = src[edge];
    int d = dst[edge];
    float coef = dinv[s] * dinv[d];
    const float4* hp = (const float4*)(h + (size_t)s * DFEAT);
    float4 v = hp[lane];                       // 32 lanes x 16B = 512B row
    float* op = out + (size_t)d * DFEAT + lane * 4;
    atomicAdd(op + 0, v.x * coef);
    atomicAdd(op + 1, v.y * coef);
    atomicAdd(op + 2, v.z * coef);
    atomicAdd(op + 3, v.w * coef);
}

// ---------------- finalize: bias + ReLU + dropout(p=0.5, deterministic hash) ----------------

__device__ __forceinline__ unsigned crd_hash(unsigned x) {
    x ^= x >> 16; x *= 0x7feb352dU;
    x ^= x >> 15; x *= 0x846ca68bU;
    x ^= x >> 16; return x;
}

__global__ void crd_finalize(float* out, const float* __restrict__ bias, int n) {
    int idx = blockIdx.x * blockDim.x + threadIdx.x;
    if (idx < n * DFEAT) {
        float v = out[idx] + bias[idx & 127];
        v = v > 0.0f ? v : 0.0f;                               // ReLU
        unsigned hsh = crd_hash((unsigned)idx ^ 0x2A6E49ABu);  // seed 42 mix
        out[idx] = (hsh & 1u) ? v * 2.0f : 0.0f;               // keep/(1-p), p=0.5
    }
}

// ---------------- launch ----------------

extern "C" void kernel_launch(void* const* d_in, const int* in_sizes, int n_in,
                              void* d_out, int out_size, void* d_ws, size_t ws_size,
                              hipStream_t stream) {
    const float* x    = (const float*)d_in[0];
    const int*   ei   = (const int*)d_in[1];
    const float* w    = (const float*)d_in[2];
    const float* bias = (const float*)d_in[3];

    const int N = in_sizes[0] / DFEAT;
    const int E = in_sizes[1] / 2;
    const int* src = ei;
    const int* dst = ei + E;

    float* out  = (float*)d_out;
    float* h    = (float*)d_ws;                    // N*128 f32 = 51.2 MB
    float* dinv = h + (size_t)N * DFEAT;           // N f32 (deg in place -> dinv)
    // Wt after dinv, 64B aligned
    size_t off = ((size_t)(N * DFEAT + N) * sizeof(float) + 63) & ~(size_t)63;
    _Float16* wT = (_Float16*)((char*)d_ws + off); // 128*128 f16 = 32 KB

    const int B = 256;
    crd_deg_init <<<(N + B - 1) / B, B, 0, stream>>>(dinv, N);
    crd_deg_edges<<<(E + B - 1) / B, B, 0, stream>>>(dst, dinv, E);
    crd_dinv     <<<(N + B - 1) / B, B, 0, stream>>>(dinv, N);

    crd_prep_w   <<<(DFEAT * DFEAT + B - 1) / B, B, 0, stream>>>(w, wT);

    const int rows_per_block = ROWTILES * TILE_ROWS;
    crd_gemm     <<<(N + rows_per_block - 1) / rows_per_block, B, 0, stream>>>(x, wT, h, N);

    crd_out_init <<<((size_t)N * DFEAT + B - 1) / B, B, 0, stream>>>(h, dinv, out, N);

    long long ethreads = (long long)E * 32;
    crd_edge_agg <<<(unsigned)((ethreads + B - 1) / B), B, 0, stream>>>(src, dst, dinv, h, out, E);

    crd_finalize <<<((size_t)N * DFEAT + B - 1) / B, B, 0, stream>>>(out, bias, N);
}
